// IngredientGNN_58239756534190
// MI455X (gfx1250) — matrix operand order repaired and checked by reading, not compiled, not run on hardware
//
#include <hip/hip_runtime.h>

typedef __attribute__((ext_vector_type(16))) _Float16 v16h;
typedef __attribute__((ext_vector_type(8)))  float    v8f;
typedef __attribute__((ext_vector_type(4)))  float    v4f;

#define TPB 256

// ---------- order-preserving float<->uint encoding for atomic max ----------
__device__ __forceinline__ unsigned enc_ord(float f) {
  unsigned u = __float_as_uint(f);
  return (u & 0x80000000u) ? ~u : (u | 0x80000000u);
}
__device__ __forceinline__ float dec_ord(unsigned u) {
  return (u & 0x80000000u) ? __uint_as_float(u & 0x7FFFFFFFu)
                           : __uint_as_float(~u);
}

// ---------- fill ----------
__global__ void k_fill_u32(unsigned* __restrict__ p, unsigned v, long long n) {
  long long i = (long long)blockIdx.x * blockDim.x + threadIdx.x;
  if (i < n) p[i] = v;
}

// ---------- WMMA GEMM: C[M,NT*16] = A[M,128] * B[128,NT*16] ----------
// fp32 in/out, f16 MACs, fp32 accumulate.
// Block = 256 threads (8 waves). Block stages the f16 B panel into LDS in the
// exact WMMA B-operand lane layout; each wave computes a 16-row strip across
// the full output width (NT accumulator tiles) so A is streamed exactly once.
template <int NT>
__global__ void k_gemm_wmma(const float* __restrict__ A, const float* __restrict__ B,
                            float* __restrict__ C, int Mtiles) {
  constexpr int K  = 128;
  constexpr int NC = NT * 16;                    // output columns
  __shared__ _Float16 ldsB[NT * 4 * 32 * 16];    // [nt][kk][lane][16 halves]

  const int t    = threadIdx.x;
  const int lane = t & 31;
  const int wave = t >> 5;
  const int half = lane >> 4;
  const int l15  = lane & 15;

  // ---- stage B panel into LDS, pre-swizzled to WMMA B layout ----
  for (int s = t; s < NT * 4 * 32; s += TPB) {
    const int nt   = s >> 7;            // s / 128
    const int rem  = s & 127;
    const int kk_i = rem >> 5;          // k-step
    const int ls   = rem & 31;          // destination lane slot
    const int col  = nt * 16 + (ls & 15);
    const int kb   = kk_i * 32 + (ls >> 4) * 16;
    v16h tmp;
#pragma unroll
    for (int j = 0; j < 16; ++j) tmp[j] = (_Float16)B[(size_t)(kb + j) * NC + col];
    *reinterpret_cast<v16h*>(&ldsB[(size_t)s * 16]) = tmp;
  }
  __syncthreads();

  const int mtile = blockIdx.x * 8 + wave;
  if (mtile >= Mtiles) return;

  v8f acc[NT];
#pragma unroll
  for (int nt = 0; nt < NT; ++nt) acc[nt] = (v8f){0.f,0.f,0.f,0.f,0.f,0.f,0.f,0.f};

  const float* arow = A + (size_t)(mtile * 16 + l15) * K;
#pragma unroll
  for (int kk = 0; kk < 4; ++kk) {
    v16h a;
    const float* p = arow + kk * 32 + half * 16;
#pragma unroll
    for (int j = 0; j < 8; ++j) { a[j] = (_Float16)p[j]; a[8 + j] = (_Float16)p[8 + j]; }
#pragma unroll
    for (int nt = 0; nt < NT; ++nt) {
      v16h b = *reinterpret_cast<const v16h*>(&ldsB[(size_t)(((nt * 4 + kk) * 32) + lane) * 16]);
      acc[nt] = __builtin_amdgcn_wmma_f32_16x16x32_f16(
          false, a, false, b, (short)0, acc[nt], false, false);
    }
  }
  // D tile: elem r -> row = half*8 + r, col = nt*16 + l15
  float* crow = C + (size_t)(mtile * 16 + half * 8) * NC + l15;
#pragma unroll
  for (int nt = 0; nt < NT; ++nt)
#pragma unroll
    for (int r = 0; r < 8; ++r) crow[(size_t)r * NC + nt * 16] = acc[nt][r];
}

// ---------- per-node attention scores: s_src/s_dst [N,H] ----------
__global__ void k_scores(const float* __restrict__ h, const float* __restrict__ a_src,
                         const float* __restrict__ a_dst, float* __restrict__ ssrc,
                         float* __restrict__ sdst, int N, int H, int C) {
  int idx = blockIdx.x * blockDim.x + threadIdx.x;
  if (idx >= N * H) return;
  int n = idx / H, hd = idx - n * H;
  const float* row = h + (size_t)n * H * C + (size_t)hd * C;
  const float* as = a_src + hd * C;
  const float* ad = a_dst + hd * C;
  float s1 = 0.f, s2 = 0.f;
  for (int c = 0; c < C; ++c) { float v = row[c]; s1 += v * as[c]; s2 += v * ad[c]; }
  ssrc[idx] = s1;
  sdst[idx] = s2;
}

__device__ __forceinline__ void edge_nodes(const int* src, const int* dst,
                                           int e, int E, int& sn, int& dn) {
  if (e < E) { sn = src[e]; dn = dst[e]; }
  else       { sn = e - E;  dn = e - E; }   // self-loops appended
}

// ---------- pass A: segment max of leaky-relu edge scores ----------
__global__ void k_edge_max(const int* __restrict__ src, const int* __restrict__ dst,
                           const float* __restrict__ ssrc, const float* __restrict__ sdst,
                           unsigned* __restrict__ emax, int E, int Etot, int H) {
  long long idx = (long long)blockIdx.x * blockDim.x + threadIdx.x;
  if (idx >= (long long)Etot * H) return;
  int e = (int)(idx / H), hd = (int)(idx - (long long)e * H);
  int sn, dn; edge_nodes(src, dst, e, E, sn, dn);
  float s = ssrc[sn * H + hd] + sdst[dn * H + hd];
  s = (s > 0.f) ? s : 0.2f * s;                       // leaky_relu, slope 0.2
  atomicMax(&emax[dn * H + hd], enc_ord(s));
}

// ---------- pass B: exp(e - max) per edge, segment-sum denominator ----------
__global__ void k_edge_exp(const int* __restrict__ src, const int* __restrict__ dst,
                           const float* __restrict__ ssrc, const float* __restrict__ sdst,
                           const unsigned* __restrict__ emax, float* __restrict__ ee,
                           float* __restrict__ denom, int E, int Etot, int H) {
  long long idx = (long long)blockIdx.x * blockDim.x + threadIdx.x;
  if (idx >= (long long)Etot * H) return;
  int e = (int)(idx / H), hd = (int)(idx - (long long)e * H);
  int sn, dn; edge_nodes(src, dst, e, E, sn, dn);
  float s = ssrc[sn * H + hd] + sdst[dn * H + hd];
  s = (s > 0.f) ? s : 0.2f * s;
  float ex = __expf(s - dec_ord(emax[dn * H + hd]));
  ee[idx] = ex;
  atomicAdd(&denom[dn * H + hd], ex);
}

// ---------- pass C: out[dst] += h[src] * alpha, float4 per thread ----------
__global__ void k_edge_agg(const int* __restrict__ src, const int* __restrict__ dst,
                           const float* __restrict__ hfeat, const float* __restrict__ ee,
                           const float* __restrict__ denom, float* __restrict__ out,
                           int E, int Etot, int H, int C) {
  const int hc  = H * C;
  const int hc4 = hc >> 2;                       // float4 groups per edge
  long long idx = (long long)blockIdx.x * blockDim.x + threadIdx.x;
  if (idx >= (long long)Etot * hc4) return;
  int e   = (int)(idx / hc4);
  int ch4 = (int)(idx - (long long)e * hc4);
  int ch  = ch4 << 2;
  int hd  = ch / C;                              // groups of 4 never cross heads
  int sn, dn; edge_nodes(src, dst, e, E, sn, dn);
  float alpha = ee[(long long)e * H + hd] / denom[dn * H + hd];
  v4f hv = *reinterpret_cast<const v4f*>(&hfeat[(size_t)sn * hc + ch]);
  float* o = &out[(size_t)dn * hc + ch];
  atomicAdd(o + 0, hv.x * alpha);
  atomicAdd(o + 1, hv.y * alpha);
  atomicAdd(o + 2, hv.z * alpha);
  atomicAdd(o + 3, hv.w * alpha);
}

// ---------- bias (+ optional relu), in place ----------
__global__ void k_bias_act(float* __restrict__ x, const float* __restrict__ b,
                           long long n, int ld, int relu) {
  long long i = (long long)blockIdx.x * blockDim.x + threadIdx.x;
  if (i >= n) return;
  float v = x[i] + b[(int)(i % ld)];
  x[i] = relu ? fmaxf(v, 0.f) : v;
}

static inline int nblk(long long n, int tpb) { return (int)((n + tpb - 1) / tpb); }

extern "C" void kernel_launch(void* const* d_in, const int* in_sizes, int n_in,
                              void* d_out, int out_size, void* d_ws, size_t ws_size,
                              hipStream_t stream) {
  const float* x     = (const float*)d_in[0];
  const int*   eidx  = (const int*)  d_in[1];
  const float* W1    = (const float*)d_in[2];
  const float* asrc1 = (const float*)d_in[3];
  const float* adst1 = (const float*)d_in[4];
  const float* bias1 = (const float*)d_in[5];
  const float* W2    = (const float*)d_in[6];
  const float* asrc2 = (const float*)d_in[7];
  const float* adst2 = (const float*)d_in[8];
  const float* bias2 = (const float*)d_in[9];

  const int N    = in_sizes[0] / 128;   // 50000
  const int E    = in_sizes[1] / 2;     // 800000
  const int Etot = E + N;               // + self loops
  const int Mt   = N / 16;              // 3125 row tiles
  const int* src = eidx;
  const int* dst = eidx + E;

  // workspace partition (floats)
  float* ws = (float*)d_ws;
  float*    h1  = ws;             ws += (size_t)N * 128;
  float*    s1s = ws;             ws += (size_t)N * 4;
  float*    s1d = ws;             ws += (size_t)N * 4;
  unsigned* em1 = (unsigned*)ws;  ws += (size_t)N * 4;
  float*    dn1 = ws;             ws += (size_t)N * 4;
  float*    ee1 = ws;             ws += (size_t)Etot * 4;
  float*    o1  = ws;             ws += (size_t)N * 128;
  float*    h2  = ws;             ws += (size_t)N * 64;
  float*    s2s = ws;             ws += (size_t)N;
  float*    s2d = ws;             ws += (size_t)N;
  unsigned* em2 = (unsigned*)ws;  ws += (size_t)N;
  float*    dn2 = ws;             ws += (size_t)N;
  float*    ee2 = ws;             ws += (size_t)Etot;
  float*    out = (float*)d_out;

  // ---- re-init accumulators every call (ws/d_out are not re-poisoned) ----
  k_fill_u32<<<nblk((long long)N * 4, TPB), TPB, 0, stream>>>(em1, 0u, (long long)N * 4);
  k_fill_u32<<<nblk((long long)N * 4, TPB), TPB, 0, stream>>>((unsigned*)dn1, 0u, (long long)N * 4);
  k_fill_u32<<<nblk((long long)N * 128, TPB), TPB, 0, stream>>>((unsigned*)o1, 0u, (long long)N * 128);
  k_fill_u32<<<nblk((long long)N, TPB), TPB, 0, stream>>>(em2, 0u, (long long)N);
  k_fill_u32<<<nblk((long long)N, TPB), TPB, 0, stream>>>((unsigned*)dn2, 0u, (long long)N);
  k_fill_u32<<<nblk((long long)N * 64, TPB), TPB, 0, stream>>>((unsigned*)out, 0u, (long long)N * 64);

  // ---- layer 1: h1 = x @ W1 (WMMA), heads=4, C=32, concat ----
  k_gemm_wmma<8><<<(Mt + 7) / 8, TPB, 0, stream>>>(x, W1, h1, Mt);
  k_scores<<<nblk((long long)N * 4, TPB), TPB, 0, stream>>>(h1, asrc1, adst1, s1s, s1d, N, 4, 32);
  k_edge_max<<<nblk((long long)Etot * 4, TPB), TPB, 0, stream>>>(src, dst, s1s, s1d, em1, E, Etot, 4);
  k_edge_exp<<<nblk((long long)Etot * 4, TPB), TPB, 0, stream>>>(src, dst, s1s, s1d, em1, ee1, dn1, E, Etot, 4);
  k_edge_agg<<<nblk((long long)Etot * 32, TPB), TPB, 0, stream>>>(src, dst, h1, ee1, dn1, o1, E, Etot, 4, 32);
  k_bias_act<<<nblk((long long)N * 128, TPB), TPB, 0, stream>>>(o1, bias1, (long long)N * 128, 128, 1);

  // ---- layer 2: h2 = relu(o1) @ W2 (WMMA), heads=1, C=64, mean==identity ----
  k_gemm_wmma<4><<<(Mt + 7) / 8, TPB, 0, stream>>>(o1, W2, h2, Mt);
  k_scores<<<nblk((long long)N, TPB), TPB, 0, stream>>>(h2, asrc2, adst2, s2s, s2d, N, 1, 64);
  k_edge_max<<<nblk((long long)Etot, TPB), TPB, 0, stream>>>(src, dst, s2s, s2d, em2, E, Etot, 1);
  k_edge_exp<<<nblk((long long)Etot, TPB), TPB, 0, stream>>>(src, dst, s2s, s2d, em2, ee2, dn2, E, Etot, 1);
  k_edge_agg<<<nblk((long long)Etot * 16, TPB), TPB, 0, stream>>>(src, dst, h2, ee2, dn2, out, E, Etot, 1, 64);
  k_bias_act<<<nblk((long long)N * 64, TPB), TPB, 0, stream>>>(out, bias2, (long long)N * 64, 64, 0);
}
